// HANForWeiboClassification_74998718923074
// MI455X (gfx1250) — compile-verified, matrix operand order checked by reading, not run on hardware
//
#include <hip/hip_runtime.h>
#include <hip/hip_bf16.h>

typedef __attribute__((ext_vector_type(16))) _Float16 v16h;
typedef __attribute__((ext_vector_type(8)))  _Float16 v8h;
typedef __attribute__((ext_vector_type(8)))  float    v8f;
typedef __attribute__((ext_vector_type(4))) unsigned int u32x4;
typedef __attribute__((ext_vector_type(8))) int i32x8;
typedef __attribute__((ext_vector_type(4))) int i32x4;

#define Nn_  1024
#define Lh_  64
#define Hh_  512
#define G4_  2048
#define Ee_  16384
#define ENe_ (Ee_ + Nn_)
#define Bb_  32
#define KC_  64      // K-chunk staged per TDM transfer (divides 256/320/512/1024/2048)

// ---------------------------------------------------------------------------
// TDM: DMA a 64-row x KC_-col f16 panel of Bt (row-major, ldbElems stride)
// into LDS at byte offset ldsOff. D# built per CDNA5 ISA 8.3/8.4:
//  group0: count=1 | lds_addr | global tile addr[56:0] | type=2
//  group1: data_size=1(2B), tensor_dim0=KC_, tensor_dim1=64,
//          tile_dim0=KC_, tile_dim1=64, tensor_dim0_stride=ldbElems
//  remaining groups: zero (2-D tile)
// ---------------------------------------------------------------------------
__device__ inline void tdm_load_panel(unsigned ldsOff, const _Float16* gptr, int ldbElems)
{
    unsigned long long ga = (unsigned long long)(size_t)gptr;
    u32x4 g0;
    g0[0] = 1u;                                               // count=1 (valid user desc)
    g0[1] = ldsOff;                                           // lds_addr (bytes)
    g0[2] = (unsigned)(ga & 0xFFFFFFFFull);                   // global_addr[31:0]
    g0[3] = (unsigned)((ga >> 32) & 0x01FFFFFFull)            // global_addr[56:32]
            | 0x80000000u;                                    // type=2 ("image")
    i32x8 g1;
    g1[0] = 0x00010000;                                       // data_size=1 -> 2 bytes
    g1[1] = (int)(KC_ << 16);                                 // tensor_dim0[15:0]
    g1[2] = (int)(64 << 16);                                  // dim0 hi=0 | tensor_dim1[15:0]
    g1[3] = (int)(KC_ << 16);                                 // dim1 hi=0 | tile_dim0
    g1[4] = 64;                                               // tile_dim1=64 | tile_dim2=0
    g1[5] = ldbElems;                                         // tensor_dim0_stride[31:0]
    g1[6] = 0;                                                // stride hi | dim1_stride lo
    g1[7] = 0;
    i32x4 z4; z4[0] = z4[1] = z4[2] = z4[3] = 0;
    i32x8 z8;
#pragma unroll
    for (int i = 0; i < 8; ++i) z8[i] = 0;
    __builtin_amdgcn_tensor_load_to_lds(g0, g1, z4, z4, z8, 0);
}

// ---------------------------------------------------------------------------
// WMMA GEMM:  C[M x Ncols](f32) = A[M x K](f16, lda) @ Bt[Ncols x K]^T (f16, ldb)
// Block = 8 waves -> 128x64 C tile. B panel double-buffered in LDS via TDM
// (wave 0 issues, TENSORcnt<=1 pipelines chunk i+1 behind chunk i's WMMAs).
// Each wave: 16x64 strip, A fragment reused across 4 WMMA tiles.
// Optional rowmap indirection for A rows, optional column bias / accumulate.
// ---------------------------------------------------------------------------
__global__ void __launch_bounds__(256)
k_wmma_gemm(const _Float16* __restrict__ A, int lda,
            const _Float16* __restrict__ Bt, int ldb,
            float* __restrict__ C, int ldc,
            int M, int Ncols, int K,
            const int* __restrict__ rowmap,
            const float* __restrict__ bias, int accum)
{
    __shared__ __align__(16) _Float16 shB[2][64 * KC_];

    int wave = threadIdx.x >> 5;
    int lane = threadIdx.x & 31;
    int strips = Ncols >> 6;
    int bm = blockIdx.x / strips;
    int tn = blockIdx.x - bm * strips;
    int tm = bm * 8 + wave;

    int half = lane >> 4;
    int lr   = lane & 15;

    int am   = tm * 16 + lr;
    int arow = rowmap ? rowmap[am] : am;
    const _Float16* ap = A  + (size_t)arow * lda + (half << 3);
    const _Float16* gB = Bt + (size_t)(tn * 64) * ldb;        // block's B panel base

    unsigned ldsBase = (unsigned)(size_t)(&shB[0][0]);
    int nch = K / KC_;

    if (wave == 0) tdm_load_panel(ldsBase, gB, ldb);          // prologue: chunk 0

    v8f acc[4];
#pragma unroll
    for (int j = 0; j < 4; ++j)
#pragma unroll
        for (int i = 0; i < 8; ++i) acc[j][i] = 0.0f;

    for (int ch = 0; ch < nch; ++ch) {
        if (wave == 0) {
            if (ch + 1 < nch) {
                tdm_load_panel(ldsBase + (unsigned)(((ch + 1) & 1) * (64 * KC_ * 2)),
                               gB + (size_t)(ch + 1) * KC_, ldb);
                __builtin_amdgcn_s_wait_tensorcnt(1);         // chunk ch resident
            } else {
                __builtin_amdgcn_s_wait_tensorcnt(0);
            }
        }
        __syncthreads();
        const _Float16* bs = &shB[ch & 1][0];
#pragma unroll
        for (int k2 = 0; k2 < KC_; k2 += 32) {
            int kk = ch * KC_ + k2;
            v8h a0 = *(const v8h*)(ap + kk);
            v8h a1 = *(const v8h*)(ap + kk + 16);
            v16h av;
#pragma unroll
            for (int i = 0; i < 8; ++i) { av[i] = a0[i]; av[i + 8] = a1[i]; }
#pragma unroll
            for (int j = 0; j < 4; ++j) {
                const _Float16* bp = bs + (size_t)(j * 16 + lr) * KC_ + k2 + (half << 3);
                v8h b0 = *(const v8h*)(bp);
                v8h b1 = *(const v8h*)(bp + 16);
                v16h bv;
#pragma unroll
                for (int i = 0; i < 8; ++i) { bv[i] = b0[i]; bv[i + 8] = b1[i]; }
                acc[j] = __builtin_amdgcn_wmma_f32_16x16x32_f16(
                    false, av, false, bv, (short)0, acc[j], false, false);
            }
        }
        __syncthreads();
    }

#pragma unroll
    for (int j = 0; j < 4; ++j) {
        int col = tn * 64 + j * 16 + lr;
        float bv = bias ? bias[col] : 0.0f;
#pragma unroll
        for (int r = 0; r < 8; ++r) {
            size_t off = (size_t)(tm * 16 + r + half * 8) * ldc + col;
            float v = acc[j][r] + bv;
            if (accum) v += C[off];
            C[off] = v;
        }
    }
}

// ---------------------------------------------------------------------------
// Elementwise / utility kernels
// ---------------------------------------------------------------------------
__global__ void k_fill_f32(float* p, float v, int n) {
    int i = blockIdx.x * blockDim.x + threadIdx.x;
    if (i < n) p[i] = v;
}
__global__ void k_fill_f16(_Float16* p, int n) {
    int i = blockIdx.x * blockDim.x + threadIdx.x;
    if (i < n) p[i] = (_Float16)0.0f;
}
__global__ void k_f32_to_f16(const float* s, _Float16* d, int n) {
    int i = blockIdx.x * blockDim.x + threadIdx.x;
    if (i < n) d[i] = (_Float16)s[i];
}
__global__ void k_acc_add(float* dst, const float* src, int n) {
    int i = blockIdx.x * blockDim.x + threadIdx.x;
    if (i < n) dst[i] += src[i];
}
// W[K x Ncols] f32 -> Wt[Ncols x Kpad] f16 (transposed, zero pad K..Kpad)
__global__ void k_conv_wT(const float* W, _Float16* Wt, int K, int Ncols, int Kpad, int n) {
    int i = blockIdx.x * blockDim.x + threadIdx.x;
    if (i >= n) return;
    int k = i % Kpad, j = i / Kpad;
    Wt[i] = (k < K) ? (_Float16)W[(size_t)k * Ncols + j] : (_Float16)0.0f;
}
// X0[(t*N+n)*320 + k] = embed[token_ids[n*L+t]*300 + k] (zero pad 300..320)
__global__ void k_embed_gather(const int* tok, const float* emb, _Float16* X0, int n) {
    int i = blockIdx.x * blockDim.x + threadIdx.x;
    if (i >= n) return;
    int k   = i % 320;
    int row = i / 320;
    int t = row >> 10, nd = row & 1023;
    float v = 0.0f;
    if (k < 300) {
        int tk = tok[nd * Lh_ + t];
        v = emb[(size_t)tk * 300 + k];
    }
    X0[i] = (_Float16)v;
}
__global__ void k_rowmap(int* rm, const int* len, int t) {
    int n = blockIdx.x * blockDim.x + threadIdx.x;
    if (n < Nn_) {
        int s = len[n] - 1 - t;
        if (s < 0) s = 0;
        rm[n] = s * Nn_ + n;
    }
}
// LSTM pointwise: gates[N x 2048] -> h,c update with mask; emit h16, optional ys
__global__ void k_lstm_point(const float* __restrict__ gates, float* h, float* c,
                             _Float16* h16, const int* len, int t,
                             _Float16* ysOut, int ysLd, int n)
{
    int i = blockIdx.x * blockDim.x + threadIdx.x;
    if (i >= n) return;
    int nd = i >> 9, k = i & 511;
    const float* g = gates + (size_t)nd * 2048;
    float gi = g[k], gf = g[512 + k], gg = g[1024 + k], go = g[1536 + k];
    float cv = c[i], hv = h[i];
    float si = 1.0f / (1.0f + __expf(-gi));
    float sf = 1.0f / (1.0f + __expf(-gf));
    float so = 1.0f / (1.0f + __expf(-go));
    float cn = sf * cv + si * tanhf(gg);
    float hn = so * tanhf(cn);
    bool m = t < len[nd];
    cv = m ? cn : cv;
    hv = m ? hn : hv;
    c[i] = cv; h[i] = hv;
    h16[i] = (_Float16)hv;
    if (ysOut) ysOut[(size_t)nd * ysLd + k] = (_Float16)hv;
}
// X1[(t*N+n)*1024 + 512 + k] = Yb[(clip(len-1-t,0)*N+n)*512 + k]
__global__ void k_yb_gather(_Float16* X1, const _Float16* Yb, const int* len, int n) {
    int i = blockIdx.x * blockDim.x + threadIdx.x;
    if (i >= n) return;
    int k  = i & 511;
    int nd = (i >> 9) & 1023;
    int t  = i >> 19;
    int s = len[nd] - 1 - t; if (s < 0) s = 0;
    X1[((size_t)t * Nn_ + nd) * 1024 + 512 + k] = Yb[((size_t)s * Nn_ + nd) * 512 + k];
}
__global__ void k_select_h0(const float* textacc, const float* denc, const float* cenc,
                            const int* ntype, _Float16* h016, int n)
{
    int i = blockIdx.x * blockDim.x + threadIdx.x;
    if (i >= n) return;
    int nd = i >> 9;
    int tp = ntype[nd];
    float v;
    if (tp == 0 || tp == 2) v = textacc[i] * 0.25f;
    else if (tp == 1)       v = denc[i];
    else                    v = cenc[i];
    h016[i] = (_Float16)v;
}

// ---------------------------------------------------------------------------
// GAT kernels
// ---------------------------------------------------------------------------
__device__ inline void atomicMaxF(float* addr, float val) {
    int* ia = (int*)addr;
    int cur = __float_as_int(*addr);
    while (__int_as_float(cur) < val) {
        int old = atomicCAS(ia, cur, __float_as_int(val));
        if (old == cur) break;
        cur = old;
    }
}
// a_s[n,hd] = dot(xt[n,hd,:], a_src[hd,:]);  a_d likewise. Wave per (n,hd).
__global__ void k_head_dots(const float* __restrict__ xt, const float* __restrict__ a_src,
                            const float* __restrict__ a_dst, float* a_s, float* a_d)
{
    int gw = (blockIdx.x * blockDim.x + threadIdx.x) >> 5;
    int lane = threadIdx.x & 31;
    if (gw >= Nn_ * 4) return;
    int n = gw >> 2, hd = gw & 3;
    const float* x  = xt + (size_t)n * 2048 + hd * 512;
    const float* as = a_src + hd * 512;
    const float* ad = a_dst + hd * 512;
    float s1 = 0.0f, s2 = 0.0f;
    for (int i = lane; i < 512; i += 32) { float xv = x[i]; s1 += xv * as[i]; s2 += xv * ad[i]; }
    for (int off = 16; off; off >>= 1) { s1 += __shfl_down(s1, off); s2 += __shfl_down(s2, off); }
    if (!lane) { a_s[gw] = s1; a_d[gw] = s2; }
}
__global__ void k_edge_logits(const int* src, const int* dst, const float* a_s, const float* a_d,
                              float* evals, float* emax, int E, int EN)
{
    int e = blockIdx.x * blockDim.x + threadIdx.x;
    if (e >= EN) return;
    int s = (e < E) ? src[e] : e - E;
    int d = (e < E) ? dst[e] : e - E;
    for (int j = 0; j < 4; ++j) {
        float v = a_s[s * 4 + j] + a_d[d * 4 + j];
        v = (v > 0.0f) ? v : 0.2f * v;         // leaky_relu(0.2)
        evals[e * 4 + j] = v;
        atomicMaxF(&emax[d * 4 + j], v);
    }
}
__global__ void k_edge_expsum(const float* evals, const float* emax, const int* dst,
                              float* ee, float* den, int E, int EN)
{
    int e = blockIdx.x * blockDim.x + threadIdx.x;
    if (e >= EN) return;
    int d = (e < E) ? dst[e] : e - E;
    for (int j = 0; j < 4; ++j) {
        float v = __expf(evals[e * 4 + j] - emax[d * 4 + j]);
        ee[e * 4 + j] = v;
        atomicAdd(&den[d * 4 + j], v);
    }
}
__global__ void k_bias_init(float* out, const float* bias, int outLd, int n) {
    int i = blockIdx.x * blockDim.x + threadIdx.x;
    if (i >= n) return;
    int nd = i >> 11, c = i & 2047;
    out[(size_t)nd * outLd + c] = bias[c];
}
// Block per edge; out[d, :] += alpha * xt[s, :]
__global__ void k_edge_aggr(const float* __restrict__ ee, const float* __restrict__ den,
                            const int* src, const int* dst, const float* __restrict__ xt,
                            float* out, int outLd, int E, int EN)
{
    int e = blockIdx.x;
    if (e >= EN) return;
    int s = (e < E) ? src[e] : e - E;
    int d = (e < E) ? dst[e] : e - E;
    for (int c = threadIdx.x; c < 2048; c += blockDim.x) {
        int j = c >> 9;
        float alpha = ee[e * 4 + j] / den[d * 4 + j];
        atomicAdd(&out[(size_t)d * outLd + c], alpha * xt[(size_t)s * 2048 + c]);
    }
}

// ---------------------------------------------------------------------------
// Semantic attention / pooling / classifier
// ---------------------------------------------------------------------------
__global__ void k_sem_score(const float* __restrict__ t1, const float* __restrict__ w2,
                            float* scores)
{
    int gw = (blockIdx.x * blockDim.x + threadIdx.x) >> 5;
    int lane = threadIdx.x & 31;
    if (gw >= 3072) return;
    const float* row = t1 + (size_t)gw * 128;
    float s = 0.0f;
    for (int i = lane; i < 128; i += 32) s += tanhf(row[i]) * w2[i];
    for (int off = 16; off; off >>= 1) s += __shfl_down(s, off);
    if (!lane) scores[gw] = s;
}
__global__ void k_sem_combine(const float* __restrict__ sems, const float* __restrict__ scores,
                              float* h, _Float16* h16, int n)
{
    int i = blockIdx.x * blockDim.x + threadIdx.x;
    if (i >= n) return;
    int nd = i >> 11, c = i & 2047;
    const float* sc = scores + nd * 3;
    float s0 = sc[0], s1 = sc[1], s2 = sc[2];
    float mx = fmaxf(s0, fmaxf(s1, s2));
    float e0 = __expf(s0 - mx), e1 = __expf(s1 - mx), e2 = __expf(s2 - mx);
    float inv = 1.0f / (e0 + e1 + e2);
    const float* z = sems + (size_t)nd * 6144;
    float v = (e0 * z[c] + e1 * z[2048 + c] + e2 * z[4096 + c]) * inv;
    h[i] = v;
    h16[i] = (_Float16)v;
}
__global__ void k_pool(const float* h, const int* batch, float* pooled, int n) {
    int i = blockIdx.x * blockDim.x + threadIdx.x;
    if (i >= n) return;
    int nd = i >> 11;
    atomicAdd(&pooled[(size_t)batch[nd] * 2048 + (i & 2047)], h[i]);
}
__global__ void k_count(const int* batch, float* cnt) {
    int n = blockIdx.x * blockDim.x + threadIdx.x;
    if (n < Nn_) atomicAdd(&cnt[batch[n]], 1.0f);
}
__global__ void k_logits(const float* pooled, const float* cnt, const float* clsW,
                         const float* clsB, float* logits)
{
    int gw = (blockIdx.x * blockDim.x + threadIdx.x) >> 5;
    int lane = threadIdx.x & 31;
    if (gw >= Bb_ * 2) return;
    int b = gw >> 1, j = gw & 1;
    float inv = 1.0f / fmaxf(cnt[b], 1.0f);
    float s = 0.0f;
    for (int i = lane; i < 2048; i += 32) s += pooled[(size_t)b * 2048 + i] * clsW[i * 2 + j];
    for (int off = 16; off; off >>= 1) s += __shfl_down(s, off);
    if (!lane) logits[b * 2 + j] = s * inv + clsB[j];
}
__global__ void k_loss(const float* logits, const int* lab, float* out0) {
    int b = threadIdx.x;
    float v = 0.0f;
    if (b < Bb_) {
        float l0 = logits[b * 2], l1 = logits[b * 2 + 1];
        float m = fmaxf(l0, l1);
        float lse = m + __logf(__expf(l0 - m) + __expf(l1 - m));
        float lp = ((lab[b] != 0) ? l1 : l0) - lse;
        v = -lp * (1.0f / (float)Bb_);
    }
    for (int off = 16; off; off >>= 1) v += __shfl_down(v, off);
    if (threadIdx.x == 0) out0[0] = v;
}

// ---------------------------------------------------------------------------
// Host orchestration
// ---------------------------------------------------------------------------
static inline int nb(long long n) { return (int)((n + 255) / 256); }

extern "C" void kernel_launch(void* const* d_in, const int* in_sizes, int n_in,
                              void* d_out, int out_size, void* d_ws, size_t ws_size,
                              hipStream_t stream)
{
    (void)in_sizes; (void)n_in; (void)out_size; (void)ws_size;

    const int*   tok   = (const int*)d_in[0];
    const int*   len   = (const int*)d_in[1];
    const float* prof  = (const float*)d_in[2];
    const int*   ntype = (const int*)d_in[3];
    const int*   eidx  = (const int*)d_in[4];
    const int*   batch = (const int*)d_in[5];
    const int*   lab   = (const int*)d_in[6];
    const float* embed = (const float*)d_in[7];
    auto F = [&](int i) { return (const float*)d_in[i]; };

    // workspace bump allocator
    char* ws = (char*)d_ws;
    size_t cur = 0;
    auto alloc = [&](size_t bytes) -> void* {
        void* p = ws + cur;
        cur = (cur + bytes + 255) & ~(size_t)255;
        return p;
    };
    _Float16* X0     = (_Float16*)alloc(65536ull * 320 * 2);
    _Float16* X1     = (_Float16*)alloc(65536ull * 1024 * 2);
    _Float16* Yb     = (_Float16*)alloc(65536ull * 512 * 2);
    _Float16* wihT   = (_Float16*)alloc(2048ull * 1024 * 2);
    _Float16* whhT   = (_Float16*)alloc(2048ull * 512 * 2);
    _Float16* wT     = (_Float16*)alloc(2048ull * 2048 * 2);
    float*    gates  = (float*)alloc(1024ull * 2048 * 4);
    float*    hbuf   = (float*)alloc(1024ull * 512 * 4);
    float*    cbuf   = (float*)alloc(1024ull * 512 * 4);
    _Float16* h16    = (_Float16*)alloc(1024ull * 512 * 2);
    float*    textacc= (float*)alloc(1024ull * 512 * 4);
    float*    denc   = (float*)alloc(1024ull * 512 * 4);
    float*    cenc   = (float*)alloc(1024ull * 512 * 4);
    _Float16* prof16 = (_Float16*)alloc(1024ull * 256 * 2);
    _Float16* h016   = (_Float16*)alloc(1024ull * 512 * 2);
    float*    hfull  = (float*)alloc(1024ull * 2048 * 4);
    _Float16* hfull16= (_Float16*)alloc(1024ull * 2048 * 2);
    float*    xt     = (float*)alloc(1024ull * 2048 * 4);
    float*    aS     = (float*)alloc(4096 * 4);
    float*    aD     = (float*)alloc(4096 * 4);
    float*    emax   = (float*)alloc(4096 * 4);
    float*    den    = (float*)alloc(4096 * 4);
    float*    evals  = (float*)alloc((size_t)ENe_ * 4 * 4);
    float*    ee     = (float*)alloc((size_t)ENe_ * 4 * 4);
    float*    sems   = (float*)alloc(1024ull * 3 * 2048 * 4);
    _Float16* z16    = (_Float16*)alloc(3072ull * 2048 * 2);
    float*    t1     = (float*)alloc(3072ull * 128 * 4);
    float*    scores = (float*)alloc(3072 * 4);
    float*    pooled = (float*)alloc(32ull * 2048 * 4);
    float*    cnt    = (float*)alloc(32 * 4);
    int*      rowmap = (int*)alloc(1024 * 4);

    auto gemm = [&](const _Float16* A, int lda, const _Float16* Bt, int ldb,
                    float* C, int ldc, int M, int Ncols, int K,
                    const int* rm, const float* bias, int accum) {
        int blocks = (M / 128) * (Ncols / 64);
        k_wmma_gemm<<<blocks, 256, 0, stream>>>(A, lda, Bt, ldb, C, ldc,
                                                M, Ncols, K, rm, bias, accum);
    };

    // ---- text encoder: embedding gather + 2-layer biLSTM --------------------
    k_embed_gather<<<nb(65536ll * 320), 256, 0, stream>>>(tok, embed, X0, 65536 * 320);
    k_fill_f32<<<nb(1024 * 512), 256, 0, stream>>>(textacc, 0.0f, 1024 * 512);

    auto lstm_dir = [&](const _Float16* X, int xld, int dinPad,
                        const float* wih, int dinReal, const float* whh, const float* b,
                        bool bwd, _Float16* ysBase, int ysLd, size_t ysSlice) {
        k_conv_wT<<<nb(2048ll * dinPad), 256, 0, stream>>>(wih, wihT, dinReal, 2048, dinPad, 2048 * dinPad);
        k_conv_wT<<<nb(2048ll * 512), 256, 0, stream>>>(whh, whhT, 512, 2048, 512, 2048 * 512);
        k_fill_f32<<<nb(1024 * 512), 256, 0, stream>>>(hbuf, 0.0f, 1024 * 512);
        k_fill_f32<<<nb(1024 * 512), 256, 0, stream>>>(cbuf, 0.0f, 1024 * 512);
        k_fill_f16<<<nb(1024 * 512), 256, 0, stream>>>(h16, 1024 * 512);
        for (int t = 0; t < Lh_; ++t) {
            const int* rm = nullptr;
            const _Float16* Ap = X;
            if (!bwd) {
                Ap = X + (size_t)t * Nn_ * xld;
            } else {
                k_rowmap<<<4, 256, 0, stream>>>(rowmap, len, t);
                rm = rowmap;
            }
            gemm(Ap, xld, wihT, dinPad, gates, 2048, 1024, 2048, dinPad, rm, b, 0);
            gemm(h16, 512, whhT, 512, gates, 2048, 1024, 2048, 512, nullptr, nullptr, 1);
            _Float16* ys = ysBase ? (ysBase + (size_t)t * ysSlice) : nullptr;
            k_lstm_point<<<nb(1024 * 512), 256, 0, stream>>>(gates, hbuf, cbuf, h16,
                                                             len, t, ys, ysLd, 1024 * 512);
        }
        k_acc_add<<<nb(1024 * 512), 256, 0, stream>>>(textacc, hbuf, 1024 * 512);
    };

    lstm_dir(X0, 320, 320, F(8),  300, F(9),  F(10), false, X1, 1024, (size_t)1024 * 1024);
    lstm_dir(X0, 320, 320, F(11), 300, F(12), F(13), true,  Yb, 512,  (size_t)1024 * 512);
    k_yb_gather<<<nb(64ll * 1024 * 512), 256, 0, stream>>>(X1, Yb, len, 64 * 1024 * 512);
    lstm_dir(X1, 1024, 1024, F(14), 1024, F(15), F(16), false, nullptr, 0, 0);
    lstm_dir(X1, 1024, 1024, F(17), 1024, F(18), F(19), true,  nullptr, 0, 0);

    // ---- data / comm encoders + node-type select ----------------------------
    k_f32_to_f16<<<nb(1024 * 256), 256, 0, stream>>>(prof, prof16, 1024 * 256);
    k_conv_wT<<<nb(512ll * 256), 256, 0, stream>>>(F(20), wT, 256, 512, 256, 512 * 256);
    gemm(prof16, 256, wT, 256, denc, 512, 1024, 512, 256, nullptr, nullptr, 0);
    k_conv_wT<<<nb(512ll * 256), 256, 0, stream>>>(F(21), wT, 256, 512, 256, 512 * 256);
    gemm(prof16, 256, wT, 256, cenc, 512, 1024, 512, 256, nullptr, nullptr, 0);
    k_select_h0<<<nb(1024 * 512), 256, 0, stream>>>(textacc, denc, cenc, ntype, h016, 1024 * 512);

    // ---- HAN: 3 layers x (3 GATs + semantic attention) ----------------------
    for (int Lyr = 0; Lyr < 3; ++Lyr) {
        int din = (Lyr == 0) ? 512 : 2048;
        const _Float16* Ain = (Lyr == 0) ? h016 : hfull16;
        int base = 22 + Lyr * 15;
        for (int m = 0; m < 3; ++m) {
            const float* gw   = F(base + m * 4 + 0);
            const float* asrc = F(base + m * 4 + 1);
            const float* adst = F(base + m * 4 + 2);
            const float* gb   = F(base + m * 4 + 3);
            const int* src = eidx + (size_t)m * 2 * Ee_;
            const int* dst = src + Ee_;
            k_conv_wT<<<nb(2048ll * din), 256, 0, stream>>>(gw, wT, din, 2048, din, 2048 * din);
            gemm(Ain, din, wT, din, xt, 2048, 1024, 2048, din, nullptr, nullptr, 0);
            k_head_dots<<<nb(4096ll * 32), 256, 0, stream>>>(xt, asrc, adst, aS, aD);
            k_fill_f32<<<nb(4096), 256, 0, stream>>>(emax, -1e30f, 4096);
            k_fill_f32<<<nb(4096), 256, 0, stream>>>(den, 0.0f, 4096);
            k_edge_logits<<<nb(ENe_), 256, 0, stream>>>(src, dst, aS, aD, evals, emax, Ee_, ENe_);
            k_edge_expsum<<<nb(ENe_), 256, 0, stream>>>(evals, emax, dst, ee, den, Ee_, ENe_);
            float* outm = sems + m * 2048;
            k_bias_init<<<nb(1024 * 2048), 256, 0, stream>>>(outm, gb, 6144, 1024 * 2048);
            k_edge_aggr<<<ENe_, 256, 0, stream>>>(ee, den, src, dst, xt, outm, 6144, Ee_, ENe_);
        }
        const float* w1 = F(base + 12);
        const float* b1 = F(base + 13);
        const float* w2 = F(base + 14);
        k_f32_to_f16<<<nb(3072ll * 2048), 256, 0, stream>>>(sems, z16, 3072 * 2048);
        k_conv_wT<<<nb(128ll * 2048), 256, 0, stream>>>(w1, wT, 2048, 128, 2048, 128 * 2048);
        gemm(z16, 2048, wT, 2048, t1, 128, 3072, 128, 2048, nullptr, b1, 0);
        k_sem_score<<<nb(3072ll * 32), 256, 0, stream>>>(t1, w2, scores);
        k_sem_combine<<<nb(1024 * 2048), 256, 0, stream>>>(sems, scores, hfull, hfull16, 1024 * 2048);
    }

    // ---- mean pool per batch + classifier + loss ----------------------------
    k_fill_f32<<<nb(32 * 2048), 256, 0, stream>>>(pooled, 0.0f, 32 * 2048);
    k_fill_f32<<<1, 256, 0, stream>>>(cnt, 0.0f, 32);
    k_pool<<<nb(1024 * 2048), 256, 0, stream>>>(hfull, batch, pooled, 1024 * 2048);
    k_count<<<4, 256, 0, stream>>>(batch, cnt);
    float* logits = (float*)d_out + 1;
    k_logits<<<nb(64ll * 32), 256, 0, stream>>>(pooled, cnt, F(67), F(68), logits);
    k_loss<<<1, 32, 0, stream>>>(logits, lab, (float*)d_out);
}
